// WanSelfAttention_75402445848880
// MI455X (gfx1250) — matrix operand order repaired and checked by reading, not compiled, not run on hardware
//
#include <hip/hip_runtime.h>
#include <hip/hip_bf16.h>

// ---------------------------------------------------------------------------
// Problem constants (match reference)
// ---------------------------------------------------------------------------
#define DIMX 1536
#define HEADS 12
#define HEAD_DIM 128
#define BATCH 2
#define SEQ 2048
#define ROWS (BATCH * SEQ)   // 4096
#define EPSV 1e-6f
#define VT_LD 40             // transposed V tile row stride (elems): 80B, 16B-aligned

typedef __attribute__((ext_vector_type(16))) __bf16 bf16x16;
typedef __attribute__((ext_vector_type(8)))  float  v8f;

union BF16Frag {
  bf16x16      v;
  __bf16       h[16];
  unsigned int u[8];
  uint4        q[2];
};

// ---------------------------------------------------------------------------
// CDNA5 async copy global->LDS (ASYNCcnt-tracked, CDNA5 ISA)
// ---------------------------------------------------------------------------
__device__ __forceinline__ void async_load_b128(unsigned lds_off, const void* gp) {
  asm volatile("global_load_async_to_lds_b128 %0, %1, off"
               :: "v"(lds_off), "v"(gp)
               : "memory");
}
#define S_WAIT_ASYNC_4 asm volatile("s_wait_asynccnt 0x4" ::: "memory")
#define S_WAIT_ASYNC_2 asm volatile("s_wait_asynccnt 0x2" ::: "memory")
#define S_WAIT_ASYNC_0 asm volatile("s_wait_asynccnt 0x0" ::: "memory")

__device__ __forceinline__ unsigned lds_addr(const void* p) {
  return (unsigned)(unsigned long long)p;   // LDS aperture: low 32 bits = LDS offset
}

// ---------------------------------------------------------------------------
// Fragment loaders matching the CDNA5 WMMA VGPR layouts (CDNA5 ISA)
// ---------------------------------------------------------------------------
// 16-bit A matrix 16x32 (MxK), row-major source with leading dim lda.
// lanes 0-15: M=lane, VGPR0-3 hold K=0..7 (pairs), VGPR4-7 hold K=16..23
// lanes 16-31: same M, VGPR0-3 hold K=8..15, VGPR4-7 hold K=24..31
__device__ __forceinline__ bf16x16 load_a_frag(const __bf16* A, int lda,
                                               int row0, int k0, int lane) {
  const int m = lane & 15, g = lane >> 4;
  const __bf16* base = A + (size_t)(row0 + m) * lda + k0;
  BF16Frag r;
#pragma unroll
  for (int v = 0; v < 8; ++v) {
    const int k = (v < 4) ? (g * 8 + 2 * v) : (16 + g * 8 + 2 * (v - 4));
    r.u[v] = *(const unsigned int*)(base + k);  // bf16 pair (k, k+1)
  }
  return r.v;
}

// 16-bit B matrix 32x16 (KxN) where the source W is stored [N, K] row-major
// (i.e. B[k][n] = W[col0+n][k0+k]).  Lane n holds column n; lanes 0-15 hold
// K=0..7,16..23, lanes 16-31 hold K=8..15,24..31 (matches A fragment).
__device__ __forceinline__ bf16x16 load_b_frag_nk(const __bf16* W, int ldw,
                                                  int col0, int k0, int lane) {
  const int n = lane & 15, g = lane >> 4;
  const __bf16* base = W + (size_t)(col0 + n) * ldw + k0 + g * 8;
  BF16Frag r;
  r.q[0] = *(const uint4*)(base);
  r.q[1] = *(const uint4*)(base + 16);
  return r.v;
}

#define WMMA_BF16(A, B, C) \
  __builtin_amdgcn_wmma_f32_16x16x32_bf16(false, (A), false, (B), (short)0, (C), false, false)

// ---------------------------------------------------------------------------
// f32 -> bf16 cast (float4-vectorized; n must be a multiple of 4)
// ---------------------------------------------------------------------------
__global__ __launch_bounds__(256) void cast_f32_bf16(const float* __restrict__ s,
                                                     __bf16* __restrict__ d, int n) {
  int i = (blockIdx.x * 256 + threadIdx.x) * 4;
  if (i < n) {
    float4 f = *(const float4*)(s + i);
    d[i + 0] = (__bf16)f.x;
    d[i + 1] = (__bf16)f.y;
    d[i + 2] = (__bf16)f.z;
    d[i + 3] = (__bf16)f.w;
  }
}

// ---------------------------------------------------------------------------
// GEMM: C[M,N] = A[M,K] (bf16) x W[N,K]^T (bf16) + bias, f32 output.
// Workgroup tile 128x128 (8 waves; wave tile 64x32 = 8 WMMAs/step).
// A/B 128x32 tiles double-buffered in LDS via async global->LDS copies.
// Grid: (M/128, N/128).  M%128==0, N%128==0, K%32==0.
// ---------------------------------------------------------------------------
__global__ __launch_bounds__(256) void gemm_bf16_f32(const __bf16* __restrict__ A,
                                                     const __bf16* __restrict__ W,
                                                     const float* __restrict__ bias,
                                                     float* __restrict__ C,
                                                     int M, int N, int K) {
  __shared__ __bf16 Abuf[2][128 * 32];   // 8 KB per buffer
  __shared__ __bf16 Bbuf[2][128 * 32];   // 8 KB per buffer
  const int tid  = threadIdx.x;
  const int lane = tid & 31;
  const int wid  = tid >> 5;
  const int wrow = wid >> 2;             // 0..1 -> 64-row slab
  const int wcol = wid & 3;              // 0..3 -> 32-col slab
  const int rowA0 = blockIdx.x * 128;
  const int colW0 = blockIdx.y * 128;
  const int n = lane & 15, g = lane >> 4;
  const int nk = K >> 5;

  // stage one 128x32 A tile + one 128x32 B tile (row-major, ld=32) into LDS:
  // 512 b128 chunks each; 2 chunks per tile per thread -> 4 async ops/thread.
  auto stage = [&](int kk, int buf) {
#pragma unroll
    for (int i = 0; i < 2; ++i) {
      const int c   = tid * 2 + i;       // 0..511
      const int row = c >> 2;            // /4
      const int col = (c & 3) * 8;
      async_load_b128(lds_addr(&Abuf[buf][c * 8]),
                      A + (size_t)(rowA0 + row) * K + kk * 32 + col);
      async_load_b128(lds_addr(&Bbuf[buf][c * 8]),
                      W + (size_t)(colW0 + row) * K + kk * 32 + col);
    }
  };

  v8f acc[4][2];
#pragma unroll
  for (int i = 0; i < 4; ++i) { acc[i][0] = (v8f){}; acc[i][1] = (v8f){}; }

  stage(0, 0);
  if (nk > 1) stage(1, 1);

  for (int kk = 0; kk < nk; ++kk) {
    // tiles complete in order: <=4 outstanding means tile kk has landed
    if (kk < nk - 1) { S_WAIT_ASYNC_4; } else { S_WAIT_ASYNC_0; }
    __syncthreads();

    const __bf16* Al = &Abuf[kk & 1][0];
    const __bf16* Bl = &Bbuf[kk & 1][0];
    bf16x16 bf0 = load_b_frag_nk(Bl, 32, wcol * 32,      0, lane);
    bf16x16 bf1 = load_b_frag_nk(Bl, 32, wcol * 32 + 16, 0, lane);
#pragma unroll
    for (int i = 0; i < 4; ++i) {
      bf16x16 af = load_a_frag(Al, 32, wrow * 64 + i * 16, 0, lane);
      acc[i][0] = WMMA_BF16(af, bf0, acc[i][0]);
      acc[i][1] = WMMA_BF16(af, bf1, acc[i][1]);
    }

    __syncthreads();                      // all waves done reading buf kk&1
    if (kk + 2 < nk) stage(kk + 2, kk & 1);
  }

  // C/D layout: VGPR v -> row v + 8*g, col = lane&15 within the 16-wide tile
#pragma unroll
  for (int j = 0; j < 2; ++j) {
    const int colb = colW0 + wcol * 32 + j * 16 + n;
    const float bv = bias[colb];
#pragma unroll
    for (int i = 0; i < 4; ++i) {
      const int rowb = rowA0 + wrow * 64 + i * 16 + 8 * g;
#pragma unroll
      for (int v = 0; v < 8; ++v)
        C[(size_t)(rowb + v) * N + colb] = acc[i][j][v] + bv;
    }
  }
}

// ---------------------------------------------------------------------------
// Per-token RMSNorm (q,k) + 3D RoPE (q,k) + bf16 convert (q,k,v).
// One workgroup of 256 threads per token row (DIMX=1536 floats).
// ---------------------------------------------------------------------------
__global__ __launch_bounds__(256) void normrope_kernel(const float* __restrict__ qf,
                                                       const float* __restrict__ kf,
                                                       const float* __restrict__ vf,
                                                       const float* __restrict__ gq,
                                                       const float* __restrict__ gk,
                                                       const float* __restrict__ fcos,
                                                       const float* __restrict__ fsin,
                                                       __bf16* __restrict__ qb,
                                                       __bf16* __restrict__ kb,
                                                       __bf16* __restrict__ vb) {
  __shared__ float red[2][8];
  const int tok = blockIdx.x;           // 0 .. ROWS-1
  const int s   = tok & (SEQ - 1);
  const int tid = threadIdx.x;
  const int lane = tid & 31, wid = tid >> 5;

  const float* qrow = qf + (size_t)tok * DIMX;
  const float* krow = kf + (size_t)tok * DIMX;
  const float* vrow = vf + (size_t)tok * DIMX;

  float sq = 0.f, sk = 0.f;
  for (int i = tid; i < DIMX; i += 256) {
    float a = qrow[i]; sq += a * a;
    float c = krow[i]; sk += c * c;
  }
#pragma unroll
  for (int off = 1; off < 32; off <<= 1) {
    sq += __shfl_xor(sq, off, 32);
    sk += __shfl_xor(sk, off, 32);
  }
  if (lane == 0) { red[0][wid] = sq; red[1][wid] = sk; }
  __syncthreads();
  float tq = 0.f, tk = 0.f;
#pragma unroll
  for (int w = 0; w < 8; ++w) { tq += red[0][w]; tk += red[1][w]; }
  const float rq = rsqrtf(tq * (1.0f / DIMX) + EPSV);
  const float rk = rsqrtf(tk * (1.0f / DIMX) + EPSV);

  // grid position: s = f*(16*16) + h*16 + w
  const int pf = s >> 8, ph = (s >> 4) & 15, pw = s & 15;

  // 12 heads * 64 pairs = 768 pairs; split [22 | 21 | 21] over (f,h,w)
  for (int p = tid; p < HEADS * 64; p += 256) {
    const int head = p >> 6, j = p & 63;
    const int pos = (j < 22) ? pf : ((j < 43) ? ph : pw);
    const float cs = fcos[pos * 64 + j];
    const float sn = fsin[pos * 64 + j];
    const int base = head * HEAD_DIM + 2 * j;
    const float qre = qrow[base] * rq * gq[base];
    const float qim = qrow[base + 1] * rq * gq[base + 1];
    const float kre = krow[base] * rk * gk[base];
    const float kim = krow[base + 1] * rk * gk[base + 1];
    const size_t o = (size_t)tok * DIMX + base;
    qb[o]     = (__bf16)(qre * cs - qim * sn);
    qb[o + 1] = (__bf16)(qre * sn + qim * cs);
    kb[o]     = (__bf16)(kre * cs - kim * sn);
    kb[o + 1] = (__bf16)(kre * sn + kim * cs);
  }
  for (int i = tid; i < DIMX; i += 256)
    vb[(size_t)tok * DIMX + i] = (__bf16)vrow[i];
}

// ---------------------------------------------------------------------------
// Flash attention: grid (SEQ/128, HEADS, BATCH); 8 waves, 16 q-rows per wave.
// K tile async-staged to LDS (rows = keys); V tile staged TRANSPOSED
// (Vt[d][key], stride VT_LD) so both K and V WMMA operands read as
// contiguous 2 x ds_load_b128 per lane.  Online softmax, all matmuls WMMA.
// ---------------------------------------------------------------------------
__global__ __launch_bounds__(256) void attn_kernel(const __bf16* __restrict__ qb,
                                                   const __bf16* __restrict__ kb,
                                                   const __bf16* __restrict__ vb,
                                                   const int* __restrict__ seq_lens,
                                                   __bf16* __restrict__ ob) {
  __shared__ __bf16 Ktile[32 * HEAD_DIM];     // 8 KB, rows = keys
  __shared__ __bf16 Vt[HEAD_DIM * VT_LD];     // 10 KB, rows = d, cols = keys
  __shared__ __bf16 Ptile[8][16 * 32];        // 1 KB per wave

  const int tid  = threadIdx.x;
  const int lane = tid & 31;
  const int wid  = tid >> 5;
  const int n = lane & 15, g = lane >> 4;
  const int head = blockIdx.y;
  const int b    = blockIdx.z;
  const int q0   = blockIdx.x * 128 + wid * 16;
  const int slen = seq_lens[b];
  const float scale = 0.088388347648318447f;  // 1/sqrt(128)

  const __bf16* Q = qb + (size_t)b * SEQ * DIMX + head * HEAD_DIM;
  const __bf16* K = kb + (size_t)b * SEQ * DIMX + head * HEAD_DIM;
  const __bf16* V = vb + (size_t)b * SEQ * DIMX + head * HEAD_DIM;

  bf16x16 qfr[4];
#pragma unroll
  for (int t = 0; t < 4; ++t) qfr[t] = load_a_frag(Q, DIMX, q0, t * 32, lane);

  v8f o[8];
#pragma unroll
  for (int t = 0; t < 8; ++t) o[t] = (v8f){};
  float mrow[8], lrow[8];
#pragma unroll
  for (int v = 0; v < 8; ++v) { mrow[v] = -3.0e38f; lrow[v] = 0.f; }

  for (int jj = 0; jj < SEQ; jj += 32) {
    __syncthreads();   // everyone done reading K/V tiles from previous block

    // K: async contiguous stage (512 b128 chunks, 2 per thread).
    // V: coalesced b128 load + transpose scatter into Vt[d][key].
#pragma unroll
    for (int i = 0; i < 2; ++i) {
      const int c    = tid * 2 + i;     // 0..511
      const int row  = c >> 4;          // key row 0..31
      const int col8 = (c & 15) * 8;    // d column 0..120
      const size_t goff = (size_t)(jj + row) * DIMX + col8;
      async_load_b128(lds_addr(&Ktile[c * 8]), K + goff);
      union { uint4 u; __bf16 h[8]; } vv;
      vv.u = *(const uint4*)(V + goff);
#pragma unroll
      for (int e = 0; e < 8; ++e)
        Vt[(col8 + e) * VT_LD + row] = vv.h[e];
    }
    S_WAIT_ASYNC_0;
    __syncthreads();

    // scores: 16 q-rows x 32 keys = two 16x16 WMMA tiles over D=128
    v8f s0 = (v8f){}, s1 = (v8f){};
#pragma unroll
    for (int t = 0; t < 4; ++t) {
      bf16x16 k0f = load_b_frag_nk(Ktile, HEAD_DIM, 0,  t * 32, lane);
      bf16x16 k1f = load_b_frag_nk(Ktile, HEAD_DIM, 16, t * 32, lane);
      s0 = WMMA_BF16(qfr[t], k0f, s0);
      s1 = WMMA_BF16(qfr[t], k1f, s1);
    }
    const bool ok0 = (jj + n) < slen;
    const bool ok1 = (jj + 16 + n) < slen;

    // online softmax; row = v + 8*g is constant across the 16-lane group
    float p0[8], p1[8];
#pragma unroll
    for (int v = 0; v < 8; ++v) {
      float a0 = ok0 ? s0[v] * scale : -3.0e38f;
      float a1 = ok1 ? s1[v] * scale : -3.0e38f;
      float mx = fmaxf(a0, a1);
#pragma unroll
      for (int off = 1; off < 16; off <<= 1) mx = fmaxf(mx, __shfl_xor(mx, off, 32));
      const float mnew  = fmaxf(mrow[v], mx);
      const float alpha = __expf(mrow[v] - mnew);
      const float e0 = ok0 ? __expf(a0 - mnew) : 0.f;
      const float e1 = ok1 ? __expf(a1 - mnew) : 0.f;
      float rs = e0 + e1;
#pragma unroll
      for (int off = 1; off < 16; off <<= 1) rs += __shfl_xor(rs, off, 32);
      lrow[v] = lrow[v] * alpha + rs;
      mrow[v] = mnew;
      p0[v] = e0;
      p1[v] = e1;
#pragma unroll
      for (int t = 0; t < 8; ++t) o[t][v] *= alpha;
    }

    // re-swizzle P from C-layout to A-layout via per-wave LDS
    __bf16* P = &Ptile[wid][0];
#pragma unroll
    for (int v = 0; v < 8; ++v) {
      const int row = v + 8 * g;
      P[row * 32 + n]      = (__bf16)p0[v];
      P[row * 32 + 16 + n] = (__bf16)p1[v];
    }
    bf16x16 pf = load_a_frag(P, 32, 0, 0, lane);

    // O += P (16x32) @ V (32x128); Vt is [d][key] so the operand reads
    // contiguous: B[k][n=d'] = Vt[d0+d'][k]
#pragma unroll
    for (int t = 0; t < 8; ++t) {
      bf16x16 vfr = load_b_frag_nk(Vt, VT_LD, t * 16, 0, lane);
      o[t] = WMMA_BF16(pf, vfr, o[t]);
    }
  }

  __bf16* O = ob + (size_t)b * SEQ * DIMX + head * HEAD_DIM;
#pragma unroll
  for (int v = 0; v < 8; ++v) {
    const float inv = (lrow[v] > 0.f) ? (1.0f / lrow[v]) : 0.f;
    const size_t r = (size_t)(q0 + v + 8 * g) * DIMX;
#pragma unroll
    for (int t = 0; t < 8; ++t)
      O[r + t * 16 + n] = (__bf16)(o[t][v] * inv);
  }
}

// ---------------------------------------------------------------------------
// Host launcher
// ---------------------------------------------------------------------------
extern "C" void kernel_launch(void* const* d_in, const int* in_sizes, int n_in,
                              void* d_out, int out_size, void* d_ws, size_t ws_size,
                              hipStream_t stream) {
  (void)in_sizes; (void)n_in; (void)out_size; (void)ws_size;

  const float* x    = (const float*)d_in[0];
  const int*   slen = (const int*)d_in[1];
  const float* fcos = (const float*)d_in[5];
  const float* fsin = (const float*)d_in[6];
  const float* wq = (const float*)d_in[7];
  const float* bq = (const float*)d_in[8];
  const float* wk = (const float*)d_in[9];
  const float* bk = (const float*)d_in[10];
  const float* wv = (const float*)d_in[11];
  const float* bv = (const float*)d_in[12];
  const float* wo = (const float*)d_in[13];
  const float* bo = (const float*)d_in[14];
  const float* gq = (const float*)d_in[15];
  const float* gk = (const float*)d_in[16];
  float* out = (float*)d_out;

  // workspace carve-out (256B aligned)
  size_t off = 0;
  char* base = (char*)d_ws;
  auto alloc = [&](size_t bytes) -> void* {
    void* p = base + off;
    off += (bytes + 255) & ~(size_t)255;
    return p;
  };
  const size_t NX = (size_t)ROWS * DIMX;       // 6,291,456 elements
  const size_t NW = (size_t)DIMX * DIMX;       // 2,359,296 elements
  __bf16* xbf = (__bf16*)alloc(NX * 2);
  __bf16* wqb = (__bf16*)alloc(NW * 2);
  __bf16* wkb = (__bf16*)alloc(NW * 2);
  __bf16* wvb = (__bf16*)alloc(NW * 2);
  __bf16* wob = (__bf16*)alloc(NW * 2);
  float*  qf  = (float*)alloc(NX * 4);
  float*  kf  = (float*)alloc(NX * 4);
  float*  vf  = (float*)alloc(NX * 4);
  __bf16* qbf = (__bf16*)alloc(NX * 2);
  __bf16* kbf = (__bf16*)alloc(NX * 2);
  __bf16* vbf = (__bf16*)alloc(NX * 2);
  __bf16* abf = (__bf16*)alloc(NX * 2);

  // 1) casts to bf16
  cast_f32_bf16<<<(int)(NX / 1024), 256, 0, stream>>>(x, xbf, (int)NX);
  cast_f32_bf16<<<(int)(NW / 1024), 256, 0, stream>>>(wq, wqb, (int)NW);
  cast_f32_bf16<<<(int)(NW / 1024), 256, 0, stream>>>(wk, wkb, (int)NW);
  cast_f32_bf16<<<(int)(NW / 1024), 256, 0, stream>>>(wv, wvb, (int)NW);
  cast_f32_bf16<<<(int)(NW / 1024), 256, 0, stream>>>(wo, wob, (int)NW);

  // 2) QKV projections (WMMA GEMM, async LDS double-buffered, f32 out)
  dim3 ggrid(ROWS / 128, DIMX / 128);
  gemm_bf16_f32<<<ggrid, 256, 0, stream>>>(xbf, wqb, bq, qf, ROWS, DIMX, DIMX);
  gemm_bf16_f32<<<ggrid, 256, 0, stream>>>(xbf, wkb, bk, kf, ROWS, DIMX, DIMX);
  gemm_bf16_f32<<<ggrid, 256, 0, stream>>>(xbf, wvb, bv, vf, ROWS, DIMX, DIMX);

  // 3) RMSNorm + RoPE + bf16
  normrope_kernel<<<ROWS, 256, 0, stream>>>(qf, kf, vf, gq, gk, fcos, fsin,
                                            qbf, kbf, vbf);

  // 4) flash attention (WMMA + async K staging + transposed V staging)
  dim3 agrid(SEQ / 128, HEADS, BATCH);
  attn_kernel<<<agrid, 256, 0, stream>>>(qbf, kbf, vbf, slen, abf);

  // 5) output projection straight into d_out (f32)
  gemm_bf16_f32<<<ggrid, 256, 0, stream>>>(abf, wob, bo, out, ROWS, DIMX, DIMX);
}
